// DSBlock_80384607912556
// MI455X (gfx1250) — matrix operand order repaired and checked by reading, not compiled, not run on hardware
//
#include <hip/hip_runtime.h>

// ---------------------------------------------------------------------------
// DSBlock on MI455X (gfx1250): every GEMM-shaped stage runs on
// V_WMMA_F32_16X16X32_F16 (f16 inputs, f32 accumulate).
// ---------------------------------------------------------------------------

typedef _Float16 v16h __attribute__((ext_vector_type(16)));
typedef _Float16 v8h  __attribute__((ext_vector_type(8)));
typedef float    v8f  __attribute__((ext_vector_type(8)));

__device__ __forceinline__ v8f wmma_f16(v16h a, v16h b, v8f c) {
    return __builtin_amdgcn_wmma_f32_16x16x32_f16(false, a, false, b,
                                                  (short)0, c, false, false);
}

// A operand: lane holds row M=lane&15; K elements {0..7,16..23} (lo lanes)
// or {8..15,24..31} (hi lanes) of a 32-K contiguous row.
__device__ __forceinline__ v16h ldA(const _Float16* row32, int hi) {
    v8h lo = *(const v8h*)(row32 + (hi ? 8 : 0));
    v8h hh = *(const v8h*)(row32 + 16 + (hi ? 8 : 0));
    v16h r;
#pragma unroll
    for (int i = 0; i < 8; ++i) { r[i] = lo[i]; r[i + 8] = hh[i]; }
    return r;
}

// B operand: lane holds column N=lane&15; K elements {0..7,16..23} (lo lanes)
// or {8..15,24..31} (hi lanes), same as ldA.
__device__ __forceinline__ v16h ldB(const _Float16* col32, int hi) {
    v8h a = *(const v8h*)(col32 + (hi ? 8 : 0));
    v8h b = *(const v8h*)(col32 + 16 + (hi ? 8 : 0));
    v16h r;
#pragma unroll
    for (int i = 0; i < 8; ++i) { r[i] = a[i]; r[i + 8] = b[i]; }
    return r;
}

// ------------------------- constants -------------------------
#define BATCH 8
#define CFULL 128
#define CHB   64           // per-branch channels
#define HH    64
#define WW    64
#define NPIX  4096         // H*W
#define SPIX  256          // 16*16 pooled
#define HEADS 4
#define HD    16

// ---------------------------------------------------------------------------
// prep: f32->f16 weight conversion + conv-weight A-operand pre-pack.
// wpack[mt][s][lane][e] mirrors the exact A-matrix lane/element layout.
// total elements = 12288 + 16384 + 36864 = 65536
// ---------------------------------------------------------------------------
__global__ void prep_kernel(const float* __restrict__ qkv_w,
                            const float* __restrict__ l_w,
                            const float* __restrict__ dep_w,
                            _Float16* __restrict__ qkv_h,
                            _Float16* __restrict__ lw_h,
                            _Float16* __restrict__ wpack) {
    int i = blockIdx.x * 256 + threadIdx.x;
    if (i < 12288) {
        qkv_h[i] = (_Float16)qkv_w[i];
    } else if (i < 12288 + 16384) {
        int j = i - 12288;
        lw_h[j] = (_Float16)l_w[j];
    } else {
        int j = i - 28672;                 // < 4*18*32*16 = 36864
        int mt = j / 9216;
        int rem = j % 9216;
        int s = rem >> 9;                  // 0..17  (tap,cin-half)
        int L = (rem >> 4) & 31;           // lane
        int e = rem & 15;                  // element within v16h
        int co = mt * 16 + (L & 15);
        int k  = (L >= 16 ? 8 : 0) + (e < 8 ? e : 16 + (e - 8));
        int cin = (s & 1) * 32 + k;
        int tap = s >> 1;                  // dy*3+dx
        wpack[j] = (_Float16)dep_w[(co * 64 + cin) * 9 + tap];
    }
}

// ---------------------------------------------------------------------------
// pad + NCHW->NHWC (f16) for the conv input (first 64 channels), 66x66 halo.
// ---------------------------------------------------------------------------
__global__ void pad_nhwc_kernel(const float* __restrict__ in,
                                _Float16* __restrict__ inh) {
    int i = blockIdx.x * 256 + threadIdx.x;
    if (i >= BATCH * 66 * 66 * 64) return;
    int c = i & 63;
    int xx = (i >> 6) % 66;
    int t  = (i >> 6) / 66;
    int yy = t % 66;
    int b  = t / 66;
    float v = 0.f;
    if (yy >= 1 && yy <= 64 && xx >= 1 && xx <= 64)
        v = in[(((size_t)b * CFULL + c) * HH + (yy - 1)) * WW + (xx - 1)];
    inh[i] = (_Float16)v;
}

// ---------------------------------------------------------------------------
// 4x4 mean pool of the second 64 channels -> Fg [B][256][64] f16 (row-major).
// ---------------------------------------------------------------------------
__global__ void pool_kernel(const float* __restrict__ in,
                            _Float16* __restrict__ Fg) {
    int i = blockIdx.x * 256 + threadIdx.x;
    if (i >= BATCH * SPIX * CHB) return;
    int c  = i & 63;
    int sx = (i >> 6) & 15;
    int sy = (i >> 10) & 15;
    int b  = i >> 14;
    float s = 0.f;
#pragma unroll
    for (int dy = 0; dy < 4; ++dy)
#pragma unroll
        for (int dx = 0; dx < 4; ++dx)
            s += in[(((size_t)b * CFULL + 64 + c) * HH + sy * 4 + dy) * WW + sx * 4 + dx];
    Fg[((size_t)b * SPIX + sy * 16 + sx) * CHB + c] = (_Float16)(s * (1.f / 16.f));
}

// ---------------------------------------------------------------------------
// 3x3 conv 64->64 as implicit GEMM: 18 WMMA k-chunks (9 taps x 2 cin halves),
// fused BN scale/bias + ReLU, output Fl [B][4096][64] f16.
// One wave per (b, y, 16-pixel x tile, 16-cout tile). 8192 waves.
// ---------------------------------------------------------------------------
__global__ void conv3x3_wmma_kernel(const _Float16* __restrict__ inh,
                                    const _Float16* __restrict__ wpack,
                                    const float* __restrict__ scale,
                                    const float* __restrict__ bias,
                                    _Float16* __restrict__ Fl) {
    int wid  = blockIdx.x * 4 + (threadIdx.x >> 5);
    int lane = threadIdx.x & 31;
    int hi   = lane >> 4;
    int mt = wid & 3;
    int xt = (wid >> 2) & 3;
    int y  = (wid >> 4) & 63;
    int b  = wid >> 10;
    int px = xt * 16 + (lane & 15);
    v8f acc = {};
#pragma unroll
    for (int s = 0; s < 18; ++s) {
        int tap = s >> 1, dy = tap / 3 - 1, dx = tap % 3 - 1, cinb = (s & 1) * 32;
        v16h A = *(const v16h*)(wpack + (size_t)((mt * 18 + s) * 32 + lane) * 16);
        const _Float16* bp =
            inh + (((size_t)(b * 66 + (y + dy + 1)) * 66) + (px + dx + 1)) * 64 + cinb;
        __builtin_prefetch(bp + 66 * 64, 0, 1);   // next output row
        v16h B = ldB(bp, hi);
        acc = wmma_f16(A, B, acc);
    }
#pragma unroll
    for (int r = 0; r < 8; ++r) {
        int co = mt * 16 + r + hi * 8;
        float v = acc[r] * scale[co] + bias[co];
        v = v > 0.f ? v : 0.f;
        Fl[((size_t)b * NPIX + y * 64 + px) * CHB + co] = (_Float16)v;
    }
}

// ---------------------------------------------------------------------------
// qkv projection: X[R,64] @ qkv_w^T[64,192] + bias, scattered into per-head
// q/k/v buffers [B][h][Nq][16] f16. One wave per (16-row, 16-col) tile.
// ---------------------------------------------------------------------------
__global__ void qkv_wmma_kernel(const _Float16* __restrict__ X, int R, int Nq,
                                const _Float16* __restrict__ Wh,
                                const float* __restrict__ bqkv,
                                _Float16* __restrict__ q,
                                _Float16* __restrict__ k,
                                _Float16* __restrict__ v) {
    int wid  = blockIdx.x * 4 + (threadIdx.x >> 5);
    int lane = threadIdx.x & 31;
    int hi   = lane >> 4;
    int ct = wid % 12;
    int rt = wid / 12;
    if (rt * 16 >= R) return;
    v8f acc = {};
#pragma unroll
    for (int kb = 0; kb < 64; kb += 32) {
        v16h A = ldA(X + (size_t)(rt * 16 + (lane & 15)) * 64 + kb, hi);
        v16h B = ldB(Wh + (size_t)(ct * 16 + (lane & 15)) * 64 + kb, hi);
        acc = wmma_f16(A, B, acc);
    }
    int c = ct * 16 + (lane & 15);
    int which = c >> 6, cc = c & 63, h = cc >> 4, d = cc & 15;
    _Float16* out = which == 0 ? q : (which == 1 ? k : v);
    float bv = bqkv[c];
#pragma unroll
    for (int r = 0; r < 8; ++r) {
        int row = rt * 16 + r + hi * 8;
        int bb = row / Nq, n = row % Nq;
        out[(((size_t)(bb * HEADS + h)) * Nq + n) * HD + d] = (_Float16)(acc[r] + bv);
    }
}

// ---------------------------------------------------------------------------
// Cross-attention (flash-style online softmax). One wave per 16-query tile
// of one (b,h). QK^T: A=q (K=16 padded to 32, scale folded), B=key rows.
// PV: P restaged via per-wave LDS slab into A layout; V transposed in LDS.
// out[(b*Nq+n)*ostride + h*16 + d] (f16) — concat buffer or Fgl buffer.
// ---------------------------------------------------------------------------
__global__ void attn_wmma_kernel(const _Float16* __restrict__ Q,
                                 const _Float16* __restrict__ K,
                                 const _Float16* __restrict__ V,
                                 int Nq, int Nk,
                                 _Float16* __restrict__ out, int ostride) {
    __shared__ __align__(32) _Float16 smem[4 * 1024];
    int wv   = threadIdx.x >> 5;
    int lane = threadIdx.x & 31;
    int hi   = lane >> 4;
    int wid  = blockIdx.x * 4 + wv;
    int qtiles = Nq >> 4;
    int qt = wid % qtiles;
    int h  = (wid / qtiles) & 3;
    int b  = wid / (qtiles * 4);
    _Float16* P  = smem + wv * 1024;        // [16 q][32 keys]
    _Float16* Vt = smem + wv * 1024 + 512;  // [16 d][32 keys]

    // A_q (scale 1/sqrt(16)=0.25 folded; K=16..31 zero pad)
    const _Float16* qrow =
        Q + (((size_t)(b * HEADS + h)) * Nq + qt * 16 + (lane & 15)) * HD;
    v8h qlo = *(const v8h*)qrow, qhi = *(const v8h*)(qrow + 8);
    v16h Aq;
#pragma unroll
    for (int i = 0; i < 8; ++i) {
        Aq[i]     = hi ? (_Float16)0.f : (_Float16)((float)qlo[i] * 0.25f);
        Aq[i + 8] = hi ? (_Float16)0.f : (_Float16)((float)qhi[i] * 0.25f);
    }

    const _Float16* Kb = K + ((size_t)(b * HEADS + h)) * Nk * HD;
    const _Float16* Vb = V + ((size_t)(b * HEADS + h)) * Nk * HD;

    float m[8], l[8];
#pragma unroll
    for (int r = 0; r < 8; ++r) { m[r] = -3.0e38f; l[r] = 0.f; }
    v8f acc = {};

    for (int kb = 0; kb < Nk; kb += 32) {
        // B operands for two 16-key subtiles (hi lanes carry the K=16..31 pad)
        v16h B0, B1;
        if (hi) {
#pragma unroll
            for (int i = 0; i < 16; ++i) { B0[i] = (_Float16)0.f; B1[i] = (_Float16)0.f; }
        } else {
            const _Float16* k0 = Kb + (size_t)(kb + (lane & 15)) * HD;
            const _Float16* k1 = Kb + (size_t)(kb + 16 + (lane & 15)) * HD;
            v8h a0 = *(const v8h*)k0, a1 = *(const v8h*)(k0 + 8);
            v8h b0 = *(const v8h*)k1, b1 = *(const v8h*)(k1 + 8);
#pragma unroll
            for (int i = 0; i < 8; ++i) {
                B0[i] = a0[i]; B0[i + 8] = a1[i];
                B1[i] = b0[i]; B1[i + 8] = b1[i];
            }
        }
        v8f z0 = {}, z1 = {};
        v8f S0 = wmma_f16(Aq, B0, z0);
        v8f S1 = wmma_f16(Aq, B1, z1);

        // stage V tile transposed: lane w owns key kb+w
        {
            const _Float16* vr = Vb + (size_t)(kb + lane) * HD;
            v8h v0 = *(const v8h*)vr, v1 = *(const v8h*)(vr + 8);
#pragma unroll
            for (int d = 0; d < 8; ++d)  Vt[d * 32 + lane] = v0[d];
#pragma unroll
            for (int d = 0; d < 8; ++d)  Vt[(d + 8) * 32 + lane] = v1[d];
        }

        // online softmax per row (N striped across 16 lanes, rows per VGPR)
#pragma unroll
        for (int r = 0; r < 8; ++r) {
            float s0 = S0[r], s1 = S1[r];
            float mx = fmaxf(s0, s1);
#pragma unroll
            for (int off = 8; off >= 1; off >>= 1)
                mx = fmaxf(mx, __shfl_xor(mx, off, 16));
            float nm = fmaxf(m[r], mx);
            float sc = __expf(m[r] - nm);
            float p0 = __expf(s0 - nm);
            float p1 = __expf(s1 - nm);
            float rs = p0 + p1;
#pragma unroll
            for (int off = 8; off >= 1; off >>= 1)
                rs += __shfl_xor(rs, off, 16);
            m[r] = nm;
            l[r] = l[r] * sc + rs;
            acc[r] = acc[r] * sc;
            int M = r + hi * 8;
            P[M * 32 + (lane & 15)]      = (_Float16)p0;
            P[M * 32 + 16 + (lane & 15)] = (_Float16)p1;
        }

        v16h Ap = ldA(P + (lane & 15) * 32, hi);
        v16h Bv = ldB(Vt + (lane & 15) * 32, hi);
        acc = wmma_f16(Ap, Bv, acc);
    }

#pragma unroll
    for (int r = 0; r < 8; ++r) {
        int n = qt * 16 + r + hi * 8;
        float v = acc[r] / l[r];
        out[((size_t)b * Nq + n) * ostride + h * HD + (lane & 15)] = (_Float16)v;
    }
}

// ---------------------------------------------------------------------------
// Bilinear 16x16 -> 64x64 (half-pixel centers) into concat channels 64..127.
// ---------------------------------------------------------------------------
__global__ void upsample_kernel(const _Float16* __restrict__ Fgl,
                                _Float16* __restrict__ concat) {
    int i = blockIdx.x * 256 + threadIdx.x;
    if (i >= BATCH * HH * WW * CHB) return;
    int c = i & 63;
    int x = (i >> 6) & 63;
    int y = (i >> 12) & 63;
    int b = i >> 18;
    float sy = y * 0.25f - 0.375f;
    float sx = x * 0.25f - 0.375f;
    float fy = floorf(sy), fx = floorf(sx);
    float wy = sy - fy,    wx = sx - fx;
    int y0 = (int)fy, x0 = (int)fx;
    int y1 = y0 + 1,  x1 = x0 + 1;
    y0 = y0 < 0 ? 0 : (y0 > 15 ? 15 : y0);
    y1 = y1 < 0 ? 0 : (y1 > 15 ? 15 : y1);
    x0 = x0 < 0 ? 0 : (x0 > 15 ? 15 : x0);
    x1 = x1 < 0 ? 0 : (x1 > 15 ? 15 : x1);
    const _Float16* base = Fgl + (size_t)b * SPIX * CHB + c;
    float g00 = (float)base[(y0 * 16 + x0) * CHB];
    float g01 = (float)base[(y0 * 16 + x1) * CHB];
    float g10 = (float)base[(y1 * 16 + x0) * CHB];
    float g11 = (float)base[(y1 * 16 + x1) * CHB];
    float v = (1.f - wy) * ((1.f - wx) * g00 + wx * g01)
            +        wy  * ((1.f - wx) * g10 + wx * g11);
    concat[((size_t)b * NPIX + y * 64 + x) * CFULL + 64 + c] = (_Float16)v;
}

// ---------------------------------------------------------------------------
// 1x1 conv 128->128: concat[pix,128] @ l_w^T, fused BN + ReLU, f32 NCHW out.
// ---------------------------------------------------------------------------
__global__ void outconv_wmma_kernel(const _Float16* __restrict__ Xc,
                                    const _Float16* __restrict__ Wh,
                                    const float* __restrict__ scale,
                                    const float* __restrict__ bias,
                                    float* __restrict__ outp) {
    int wid  = blockIdx.x * 4 + (threadIdx.x >> 5);
    int lane = threadIdx.x & 31;
    int hi   = lane >> 4;
    int ct = wid & 7;
    int rt = wid >> 3;
    v8f acc = {};
#pragma unroll
    for (int kb = 0; kb < 128; kb += 32) {
        v16h A = ldA(Xc + (size_t)(rt * 16 + (lane & 15)) * CFULL + kb, hi);
        v16h B = ldB(Wh + (size_t)(ct * 16 + (lane & 15)) * CFULL + kb, hi);
        acc = wmma_f16(A, B, acc);
    }
    int c = ct * 16 + (lane & 15);
    float sc = scale[c], bi = bias[c];
#pragma unroll
    for (int r = 0; r < 8; ++r) {
        int row = rt * 16 + r + hi * 8;
        int bb = row >> 12, pix = row & 4095;
        float v = acc[r] * sc + bi;
        v = v > 0.f ? v : 0.f;
        outp[(((size_t)bb * CFULL + c) << 12) + pix] = v;
    }
}

// ---------------------------------------------------------------------------
extern "C" void kernel_launch(void* const* d_in, const int* in_sizes, int n_in,
                              void* d_out, int out_size, void* d_ws, size_t ws_size,
                              hipStream_t stream) {
    (void)in_sizes; (void)n_in; (void)out_size; (void)ws_size;
    const float* inp   = (const float*)d_in[0];
    const float* dep_w = (const float*)d_in[1];
    const float* dep_s = (const float*)d_in[2];
    const float* dep_b = (const float*)d_in[3];
    const float* qkvw  = (const float*)d_in[4];
    const float* qkvb  = (const float*)d_in[5];
    const float* lw    = (const float*)d_in[6];
    const float* ls    = (const float*)d_in[7];
    const float* lb    = (const float*)d_in[8];
    float* out = (float*)d_out;

    char* p = (char*)d_ws;
    auto alloc = [&](size_t halves) -> _Float16* {
        _Float16* r = (_Float16*)p;
        p += ((halves * 2 + 255) / 256) * 256;
        return r;
    };
    _Float16* qkv_h = alloc(12288);                       // [192][64]
    _Float16* lw_h  = alloc(16384);                       // [128][128]
    _Float16* wpack = alloc(4 * 18 * 32 * 16);            // conv A pre-pack
    _Float16* inh   = alloc((size_t)BATCH * 66 * 66 * 64);// padded NHWC
    _Float16* Fl    = alloc((size_t)BATCH * NPIX * CHB);  // local feat
    _Float16* Fg    = alloc((size_t)BATCH * SPIX * CHB);  // pooled feat
    _Float16* ql = alloc((size_t)BATCH * HEADS * NPIX * HD);
    _Float16* kl = alloc((size_t)BATCH * HEADS * NPIX * HD);
    _Float16* vl = alloc((size_t)BATCH * HEADS * NPIX * HD);
    _Float16* qg = alloc((size_t)BATCH * HEADS * SPIX * HD);
    _Float16* kg = alloc((size_t)BATCH * HEADS * SPIX * HD);
    _Float16* vg = alloc((size_t)BATCH * HEADS * SPIX * HD);
    _Float16* cc = alloc((size_t)BATCH * NPIX * CFULL);   // concat feat f16
    _Float16* Fgl = alloc((size_t)BATCH * SPIX * CHB);    // pre-upsample

    prep_kernel<<<256, 256, 0, stream>>>(qkvw, lw, dep_w, qkv_h, lw_h, wpack);
    pad_nhwc_kernel<<<(BATCH * 66 * 66 * 64 + 255) / 256, 256, 0, stream>>>(inp, inh);
    pool_kernel<<<(BATCH * SPIX * CHB + 255) / 256, 256, 0, stream>>>(inp, Fg);
    conv3x3_wmma_kernel<<<2048, 128, 0, stream>>>(inh, wpack, dep_s, dep_b, Fl);
    // local qkv: 32768 rows -> 2048 row tiles * 12 col tiles = 24576 waves
    qkv_wmma_kernel<<<6144, 128, 0, stream>>>(Fl, BATCH * NPIX, NPIX, qkv_h, qkvb, ql, kl, vl);
    // global qkv: 2048 rows -> 128 * 12 = 1536 waves
    qkv_wmma_kernel<<<384, 128, 0, stream>>>(Fg, BATCH * SPIX, SPIX, qkv_h, qkvb, qg, kg, vg);
    // attn_lg: local queries (4096) vs global kv (256) -> concat ch 0..63
    attn_wmma_kernel<<<2048, 128, 0, stream>>>(ql, kg, vg, NPIX, SPIX, cc, CFULL);
    // attn_gl: global queries (256) vs local kv (4096) -> Fgl
    attn_wmma_kernel<<<128, 128, 0, stream>>>(qg, kl, vl, SPIX, NPIX, Fgl, CHB);
    upsample_kernel<<<(BATCH * HH * WW * CHB + 255) / 256, 256, 0, stream>>>(Fgl, cc);
    outconv_wmma_kernel<<<4096, 128, 0, stream>>>(cc, lw_h, ls, lb, out);
}
